// LSTMEncDec_9972914061571
// MI455X (gfx1250) — compile-verified, matrix operand order checked by reading, not run on hardware
//
#include <hip/hip_runtime.h>
#include <hip/hip_bf16.h>

// ---------------- problem constants ----------------
#define B_    2048
#define T_    100
#define D_    51
#define H_    128
#define GATES 512          // 4*H
#define BM    16           // batch rows per workgroup (one WMMA M tile)
#define NWAVES 16
#define BLOCK  (NWAVES * 32)   // 512 threads -> 4 waves per SIMD32 resident
#define NTW   2            // gate n-tiles per wave (16 waves * 2 * 16 = 512 cols)

typedef __bf16 bf16t;
typedef __attribute__((ext_vector_type(16))) __bf16 v16bf;
typedef __attribute__((ext_vector_type(8)))  float  v8f;

union Frag {
    v16bf v;
    uint4 q[2];
    __bf16 h[16];
};

__device__ __forceinline__ float sigmoidf_(float x) {
    return 1.0f / (1.0f + __expf(-x));
}
__device__ __forceinline__ float tanhf_(float x) {
    x = fminf(fmaxf(x, -15.0f), 15.0f);
    float e = __expf(2.0f * x);
    return (e - 1.0f) / (e + 1.0f);
}

// Add an opaque zero offset to a pointer: keeps pointer provenance (loads stay
// global_load_*, LOADcnt only) while making the address loop-variant so weight
// fragment loads can't be hoisted out of the recurrence loop and spilled.
template <typename T>
__device__ __forceinline__ const T* launder_off(const T* p) {
    size_t o = 0;
    asm volatile("" : "+s"(o));
    return p + o;
}

// ---------------- fragment layouts (CDNA5 ISA 7.12.2, wave32) ----------
// 16-bit A (16x32): lane l -> m = l&15, hs = l>>4; element e -> k =
//   kt*32 + (e>>3)*16 + hs*8 + (e&7)
// B fragments are pre-shuffled into the same per-lane order by prep kernel,
// stored contiguously: elem_off = ((kt*Ntiles + nt)*32 + lane)*16 + e.

__device__ __forceinline__ Frag load_bfrag(const bf16t* base,
                                           int kt, int nt, int Ntiles, int lane) {
    Frag f;
    const uint4* p = (const uint4*)(base + (((size_t)(kt * Ntiles + nt)) * 32 + lane) * 16);
    f.q[0] = p[0];
    f.q[1] = p[1];
    return f;
}

__device__ __forceinline__ Frag load_afrag_lds(const bf16t* __restrict__ lds,
                                               int kt, int lane) {
    Frag f;
    const uint4* p = (const uint4*)(lds + (size_t)(kt * 32 + lane) * 16);
    f.q[0] = p[0];
    f.q[1] = p[1];
    return f;
}

// A fragment from bf16 row-major [rows x rowstride] (k contiguous)
__device__ __forceinline__ Frag load_afrag_bf16(const bf16t* __restrict__ x,
                                                size_t row_stride, int rowbase,
                                                int kt, int lane) {
    Frag f;
    const int m  = lane & 15;
    const int hs = lane >> 4;
    const bf16t* p = x + (size_t)(rowbase + m) * row_stride + kt * 32 + hs * 8;
    f.q[0] = *(const uint4*)(p);
    f.q[1] = *(const uint4*)(p + 16);
    return f;
}

// A fragment from f32 row-major with K bound (zero-pad past Kcols)
__device__ __forceinline__ Frag load_afrag_f32(const float* __restrict__ x,
                                               size_t row_stride, int rowbase,
                                               int kt, int Kcols, int lane) {
    Frag f;
    const int m  = lane & 15;
    const int hs = lane >> 4;
    const float* p = x + (size_t)(rowbase + m) * row_stride;
#pragma unroll
    for (int e = 0; e < 16; ++e) {
        int k = kt * 32 + ((e >> 3) << 4) + (hs << 3) + (e & 7);
        f.h[e] = (bf16t)((k < Kcols) ? p[k] : 0.0f);
    }
    return f;
}

// scatter one h value (row m, col j in [0,128)) into A-fragment order in LDS
__device__ __forceinline__ void store_h_frag(bf16t* __restrict__ lds,
                                             int m, int j, float hval) {
    const int kt = j >> 5;
    const int r  = j & 31;
    const int e  = ((r >> 4) << 3) | (r & 7);
    const int hs = (r >> 3) & 1;
    const int l  = (hs << 4) | m;
    lds[(size_t)(kt * 32 + l) * 16 + e] = (bf16t)hval;
}

// ---------------- weight reorder: f32 [Nrows x Kcols] -> bf16 B fragments ----
__global__ void prep_frag_kernel(const float* __restrict__ W, bf16t* __restrict__ dst,
                                 int Nrows, int Kcols, int Kp, int Ntiles) {
    const int total = (Kp / 32) * Ntiles * 32 * 16;
    int tid = blockIdx.x * blockDim.x + threadIdx.x;
    if (tid >= total) return;
    const int e    = tid & 15;
    const int l    = (tid >> 4) & 31;
    const int frag = tid >> 9;
    const int kt = frag / Ntiles;
    const int nt = frag % Ntiles;
    const int n  = nt * 16 + (l & 15);
    const int hs = l >> 4;
    const int k  = kt * 32 + ((e >> 3) << 4) + (hs << 3) + (e & 7);
    float v = 0.0f;
    if (n < Nrows && k < Kcols) v = W[(size_t)n * Kcols + k];
    dst[tid] = (bf16t)v;
}

// ---------------- one encoder LSTM layer, persistent over T ----------------
// MODE 0: x = f32 src [B,T,D] (K padded 51->64).  MODE 1: x = bf16 hs [T,B,H].
template <int MODE>
__global__ __launch_bounds__(BLOCK) void lstm_layer_kernel(
    const float* __restrict__ src, const bf16t* __restrict__ xs,
    const bf16t* __restrict__ wih_frag, const bf16t* __restrict__ whh_frag,
    const float* __restrict__ bias, bf16t* __restrict__ hs_out,
    float* __restrict__ hT_out, float* __restrict__ cT_out) {
    __shared__ float gates[BM][GATES];       // 32 KB
    __shared__ bf16t hfrag[4 * 32 * 16];     // 4 KB, A-fragment order

    const int tid  = threadIdx.x;
    const int lane = tid & 31;
    const int wv   = tid >> 5;
    const int bm   = blockIdx.x * BM;
    const int em   = tid >> 5;         // cell-update row 0..15
    const int ej   = (tid & 31) * 4;   // cell-update col base (4 cols/thread)

    for (int i = tid; i < 4 * 32 * 16; i += BLOCK) hfrag[i] = (bf16t)0.0f;

    float c[4];
#pragma unroll
    for (int r = 0; r < 4; ++r) c[r] = 0.0f;

    float breg[NTW];
#pragma unroll
    for (int i = 0; i < NTW; ++i) breg[i] = bias[(wv * NTW + i) * 16 + (lane & 15)];

    constexpr int KX = (MODE == 0) ? 2 : 4;
    __syncthreads();

    for (int t = 0; t < T_; ++t) {
        // per-iteration opaque offsets: keep weight-fragment loads inside the
        // loop (streamed from L2 as global_load_b128), never hoisted/spilled
        const bf16t* wx = launder_off(wih_frag);
        const bf16t* wh = launder_off(whh_frag);

        // ---- phase 1: gates = x@WihT + h@WhhT + b (WMMA) ----
        Frag ax[KX], ah[4];
#pragma unroll
        for (int kt = 0; kt < KX; ++kt) {
            if (MODE == 0)
                ax[kt] = load_afrag_f32(src + (size_t)t * D_, (size_t)T_ * D_, bm, kt, D_, lane);
            else
                ax[kt] = load_afrag_bf16(xs + (size_t)t * B_ * H_, H_, bm, kt, lane);
        }
#pragma unroll
        for (int kt = 0; kt < 4; ++kt) ah[kt] = load_afrag_lds(hfrag, kt, lane);

        // warm L2/L0 for next step's activation rows (global_prefetch_b8)
        if (t + 1 < T_) {
            if (MODE == 0)
                __builtin_prefetch(src + (size_t)(bm + (lane & 15)) * T_ * D_ + (size_t)(t + 1) * D_, 0, 3);
            else
                __builtin_prefetch(xs + ((size_t)(t + 1) * B_ + bm + (lane & 15)) * H_, 0, 3);
        }

#pragma unroll
        for (int i = 0; i < NTW; ++i) {
            const int nt = wv * NTW + i;
            v8f acc;
#pragma unroll
            for (int r = 0; r < 8; ++r) acc[r] = breg[i];
#pragma unroll
            for (int kt = 0; kt < KX; ++kt) {
                Frag b = load_bfrag(wx, kt, nt, 32, lane);
                acc = __builtin_amdgcn_wmma_f32_16x16x32_bf16(false, ax[kt].v, false, b.v,
                                                              (short)0, acc, false, false);
            }
#pragma unroll
            for (int kt = 0; kt < 4; ++kt) {
                Frag b = load_bfrag(wh, kt, nt, 32, lane);
                acc = __builtin_amdgcn_wmma_f32_16x16x32_bf16(false, ah[kt].v, false, b.v,
                                                              (short)0, acc, false, false);
            }
            const int n  = nt * 16 + (lane & 15);
            const int hs = lane >> 4;
#pragma unroll
            for (int r = 0; r < 8; ++r) gates[r + 8 * hs][n] = acc[r];
        }
        __syncthreads();

        // ---- phase 2: fused LSTM cell, h -> global + LDS frag ----
        float hv[4];
#pragma unroll
        for (int r = 0; r < 4; ++r) {
            const int j = ej + r;
            const float gi = gates[em][j];
            const float gf = gates[em][H_ + j];
            const float gg = gates[em][2 * H_ + j];
            const float go = gates[em][3 * H_ + j];
            const float cn = sigmoidf_(gf) * c[r] + sigmoidf_(gi) * tanhf_(gg);
            c[r]  = cn;
            hv[r] = sigmoidf_(go) * tanhf_(cn);
        }
        {
            bf16t hb[4];
#pragma unroll
            for (int r = 0; r < 4; ++r) hb[r] = (bf16t)hv[r];
            *(uint2*)(hs_out + ((size_t)t * B_ + bm + em) * H_ + ej) = *(uint2*)hb;
#pragma unroll
            for (int r = 0; r < 4; ++r) store_h_frag(hfrag, em, ej + r, hv[r]);
        }
        if (t == T_ - 1) {
#pragma unroll
            for (int r = 0; r < 4; ++r) {
                hT_out[(size_t)(bm + em) * H_ + ej + r] = hv[r];
                cT_out[(size_t)(bm + em) * H_ + ej + r] = c[r];
            }
        }
        __syncthreads();
    }
}

// ---------------- decoder layer cell (gates + state update) ----------------
template <int KX>
__device__ __forceinline__ void dec_layer(
    const bf16t* __restrict__ xf, const bf16t* __restrict__ hf,
    const bf16t* wxf, const bf16t* whf,
    const float* breg, float* c, float (*gates)[GATES], bf16t* __restrict__ hf_out,
    int lane, int wv, int em, int ej) {
    Frag ax[KX], ah[4];
#pragma unroll
    for (int kt = 0; kt < KX; ++kt) ax[kt] = load_afrag_lds(xf, kt, lane);
#pragma unroll
    for (int kt = 0; kt < 4; ++kt) ah[kt] = load_afrag_lds(hf, kt, lane);

#pragma unroll
    for (int i = 0; i < NTW; ++i) {
        const int nt = wv * NTW + i;
        v8f acc;
#pragma unroll
        for (int r = 0; r < 8; ++r) acc[r] = breg[i];
#pragma unroll
        for (int kt = 0; kt < KX; ++kt) {
            Frag b = load_bfrag(wxf, kt, nt, 32, lane);
            acc = __builtin_amdgcn_wmma_f32_16x16x32_bf16(false, ax[kt].v, false, b.v,
                                                          (short)0, acc, false, false);
        }
#pragma unroll
        for (int kt = 0; kt < 4; ++kt) {
            Frag b = load_bfrag(whf, kt, nt, 32, lane);
            acc = __builtin_amdgcn_wmma_f32_16x16x32_bf16(false, ah[kt].v, false, b.v,
                                                          (short)0, acc, false, false);
        }
        const int n  = nt * 16 + (lane & 15);
        const int hs = lane >> 4;
#pragma unroll
        for (int r = 0; r < 8; ++r) gates[r + 8 * hs][n] = acc[r];
    }
    __syncthreads();
#pragma unroll
    for (int r = 0; r < 4; ++r) {
        const int j = ej + r;
        const float gi = gates[em][j];
        const float gf = gates[em][H_ + j];
        const float gg = gates[em][2 * H_ + j];
        const float go = gates[em][3 * H_ + j];
        const float cn = sigmoidf_(gf) * c[r] + sigmoidf_(gi) * tanhf_(gg);
        c[r] = cn;
        store_h_frag(hf_out, em, j, sigmoidf_(go) * tanhf_(cn));
    }
    __syncthreads();
}

// ---------------- autoregressive decoder, persistent over T ----------------
__global__ __launch_bounds__(BLOCK) void lstm_decoder_kernel(
    const bf16t* __restrict__ w0x, const bf16t* __restrict__ w0h, const float* __restrict__ b0,
    const bf16t* __restrict__ w1x, const bf16t* __restrict__ w1h, const float* __restrict__ b1,
    const bf16t* __restrict__ w2x, const bf16t* __restrict__ w2h, const float* __restrict__ b2,
    const bf16t* __restrict__ wff, const float* __restrict__ fbias,
    const float* __restrict__ hT, const float* __restrict__ cT,  // [3,B,H] f32
    float* __restrict__ out) {                                   // [B,T,D] f32
    __shared__ float gates[BM][GATES];        // 32 KB (reused across layers)
    __shared__ bf16t hfrag[3][4 * 32 * 16];   // 12 KB
    __shared__ bf16t xfrag[2 * 32 * 16];      // 2 KB (D padded to 64)

    const int tid  = threadIdx.x;
    const int lane = tid & 31;
    const int wv   = tid >> 5;
    const int bm   = blockIdx.x * BM;
    const int em   = tid >> 5;
    const int ej   = (tid & 31) * 4;

    for (int i = tid; i < 2 * 32 * 16; i += BLOCK) xfrag[i] = (bf16t)0.0f;  // x0 = zeros + pad

    float c[3][4];
#pragma unroll
    for (int L = 0; L < 3; ++L) {
#pragma unroll
        for (int r = 0; r < 4; ++r) {
            const int j = ej + r;
            const size_t idx = ((size_t)L * B_ + bm + em) * H_ + j;
            c[L][r] = cT[idx];
            store_h_frag(hfrag[L], em, j, hT[idx]);
        }
    }

    float breg[3][NTW];
    const float* bs[3] = {b0, b1, b2};
#pragma unroll
    for (int L = 0; L < 3; ++L)
#pragma unroll
        for (int i = 0; i < NTW; ++i) breg[L][i] = bs[L][(wv * NTW + i) * 16 + (lane & 15)];

    float fbreg[NTW] = {0.0f, 0.0f};
    if (wv < 2) {
#pragma unroll
        for (int i = 0; i < NTW; ++i) {
            const int col = (wv * NTW + i) * 16 + (lane & 15);
            if (col < D_) fbreg[i] = fbias[col];
        }
    }
    __syncthreads();

    for (int t = 0; t < T_; ++t) {
        // opaque per-iteration weight offsets: stream fragments from L2 as
        // global loads, never hoist/spill across the autoregressive loop
        const bf16t* p0x = launder_off(w0x);
        const bf16t* p0h = launder_off(w0h);
        const bf16t* p1x = launder_off(w1x);
        const bf16t* p1h = launder_off(w1h);
        const bf16t* p2x = launder_off(w2x);
        const bf16t* p2h = launder_off(w2h);
        const bf16t* pff = launder_off(wff);

        dec_layer<2>(xfrag,    hfrag[0], p0x, p0h, breg[0], c[0], gates, hfrag[0], lane, wv, em, ej);
        dec_layer<4>(hfrag[0], hfrag[1], p1x, p1h, breg[1], c[1], gates, hfrag[1], lane, wv, em, ej);
        dec_layer<4>(hfrag[1], hfrag[2], p2x, p2h, breg[2], c[2], gates, hfrag[2], lane, wv, em, ej);

        // projection: pred = h2 @ fW^T + fb  (waves 0..1, N = 64 covering D=51)
        Frag a2[4];
#pragma unroll
        for (int kt = 0; kt < 4; ++kt) a2[kt] = load_afrag_lds(hfrag[2], kt, lane);
        if (wv < 2) {
#pragma unroll
            for (int i = 0; i < NTW; ++i) {
                v8f acc;
#pragma unroll
                for (int r = 0; r < 8; ++r) acc[r] = fbreg[i];
#pragma unroll
                for (int kt = 0; kt < 4; ++kt) {
                    Frag b = load_bfrag(pff, kt, wv * NTW + i, 4, lane);
                    acc = __builtin_amdgcn_wmma_f32_16x16x32_bf16(false, a2[kt].v, false, b.v,
                                                                  (short)0, acc, false, false);
                }
                const int col = (wv * NTW + i) * 16 + (lane & 15);
                const int hs  = lane >> 4;
                if (col < D_) {
#pragma unroll
                    for (int r = 0; r < 8; ++r) {
                        const int m = r + 8 * hs;
                        out[((size_t)(bm + m) * T_ + (T_ - 1 - t)) * D_ + col] = acc[r];  // reversed t
                        store_h_frag(xfrag, m, col, acc[r]);  // feedback x_{t+1} = pred_t
                    }
                }
            }
        }
        __syncthreads();
    }
}

// ---------------- host side ----------------
extern "C" void kernel_launch(void* const* d_in, const int* in_sizes, int n_in,
                              void* d_out, int out_size, void* d_ws, size_t ws_size,
                              hipStream_t stream) {
    (void)in_sizes; (void)n_in; (void)out_size; (void)ws_size;
    const float* src = (const float*)d_in[0];
    const float* eW[3] = {(const float*)d_in[1], (const float*)d_in[4], (const float*)d_in[7]};
    const float* eU[3] = {(const float*)d_in[2], (const float*)d_in[5], (const float*)d_in[8]};
    const float* eb[3] = {(const float*)d_in[3], (const float*)d_in[6], (const float*)d_in[9]};
    const float* dW[3] = {(const float*)d_in[10], (const float*)d_in[13], (const float*)d_in[16]};
    const float* dU[3] = {(const float*)d_in[11], (const float*)d_in[14], (const float*)d_in[17]};
    const float* db[3] = {(const float*)d_in[12], (const float*)d_in[15], (const float*)d_in[18]};
    const float* fW = (const float*)d_in[19];
    const float* fb = (const float*)d_in[20];

    char* ws = (char*)d_ws;
    size_t off = 0;
    auto take = [&](size_t bytes) -> char* {
        char* p = ws + off;
        off = (off + bytes + 255) & ~(size_t)255;
        return p;
    };

    bf16t *eWf[3], *eUf[3], *dWf[3], *dUf[3];
    eWf[0] = (bf16t*)take((size_t)64 * 512 * 2);
    eWf[1] = (bf16t*)take((size_t)128 * 512 * 2);
    eWf[2] = (bf16t*)take((size_t)128 * 512 * 2);
    for (int l = 0; l < 3; ++l) eUf[l] = (bf16t*)take((size_t)128 * 512 * 2);
    dWf[0] = (bf16t*)take((size_t)64 * 512 * 2);
    dWf[1] = (bf16t*)take((size_t)128 * 512 * 2);
    dWf[2] = (bf16t*)take((size_t)128 * 512 * 2);
    for (int l = 0; l < 3; ++l) dUf[l] = (bf16t*)take((size_t)128 * 512 * 2);
    bf16t* fWf = (bf16t*)take((size_t)128 * 64 * 2);
    bf16t* hs  = (bf16t*)take((size_t)T_ * B_ * H_ * 2);
    float* hT  = (float*)take((size_t)3 * B_ * H_ * 4);
    float* cTp = (float*)take((size_t)3 * B_ * H_ * 4);

    auto prep = [&](const float* W, bf16t* dst, int Nrows, int Kcols, int Kp, int Ntiles) {
        int total = (Kp / 32) * Ntiles * 32 * 16;
        prep_frag_kernel<<<(total + 255) / 256, 256, 0, stream>>>(W, dst, Nrows, Kcols, Kp, Ntiles);
    };
    prep(eW[0], eWf[0], 512, D_, 64, 32);
    prep(eW[1], eWf[1], 512, H_, 128, 32);
    prep(eW[2], eWf[2], 512, H_, 128, 32);
    for (int l = 0; l < 3; ++l) prep(eU[l], eUf[l], 512, H_, 128, 32);
    prep(dW[0], dWf[0], 512, D_, 64, 32);
    prep(dW[1], dWf[1], 512, H_, 128, 32);
    prep(dW[2], dWf[2], 512, H_, 128, 32);
    for (int l = 0; l < 3; ++l) prep(dU[l], dUf[l], 512, H_, 128, 32);
    prep(fW, fWf, D_, H_, 128, 4);

    const dim3 grid(B_ / BM), block(BLOCK);
    lstm_layer_kernel<0><<<grid, block, 0, stream>>>(src, nullptr, eWf[0], eUf[0], eb[0],
                                                     hs, hT, cTp);
    lstm_layer_kernel<1><<<grid, block, 0, stream>>>(nullptr, hs, eWf[1], eUf[1], eb[1],
                                                     hs, hT + (size_t)B_ * H_, cTp + (size_t)B_ * H_);
    lstm_layer_kernel<1><<<grid, block, 0, stream>>>(nullptr, hs, eWf[2], eUf[2], eb[2],
                                                     hs, hT + (size_t)2 * B_ * H_, cTp + (size_t)2 * B_ * H_);

    lstm_decoder_kernel<<<grid, block, 0, stream>>>(dWf[0], dUf[0], db[0],
                                                    dWf[1], dUf[1], db[1],
                                                    dWf[2], dUf[2], db[2],
                                                    fWf, fb, hT, cTp, (float*)d_out);
}
